// GroupRangeConvAttention_19224273617305
// MI455X (gfx1250) — compile-verified, matrix operand order checked by reading, not run on hardware
//
#include <hip/hip_runtime.h>
#include <hip/hip_bf16.h>

// Problem constants
#define BT 192   // B*T
#define NN 2048  // N
#define DD 64    // D
#define RR 16    // R
#define GG 128   // G = N/R
#define HD 128   // H*D

typedef __bf16 bf16_t;
typedef __attribute__((ext_vector_type(16))) __bf16 v16bf;
typedef __attribute__((ext_vector_type(8)))  __bf16 v8bf;
typedef __attribute__((ext_vector_type(4)))  __bf16 v4bf;
typedef __attribute__((ext_vector_type(8)))  float  v8f;

struct PtrsF4 { const float* p[4]; };
struct PtrsI4 { const int*   p[4]; };

// ---------- WMMA fragment helpers (CDNA5 ISA 7.12.2 layouts, wave32) ----------
__device__ __forceinline__ v8f v8f_zero() {
  v8f z;
#pragma unroll
  for (int i = 0; i < 8; ++i) z[i] = 0.0f;
  return z;
}

// A-matrix 16x32 bf16 fragment from a row-major [16][ld] tile.
// Lane L: row m=L%16, K-halves at {8h..8h+7} and {16+8h..16+8h+7} -> two b128 loads.
__device__ __forceinline__ v16bf frag_a_ld(const bf16_t* base, int ld) {
  const int lane = threadIdx.x & 31;
  const int m = lane & 15, half = lane >> 4;
  const bf16_t* p = base + m * ld + (half << 3);
  v8bf lo = *(const v8bf*)p;
  v8bf hi = *(const v8bf*)(p + 16);
  return __builtin_shufflevector(lo, hi, 0, 1, 2, 3, 4, 5, 6, 7,
                                 8, 9, 10, 11, 12, 13, 14, 15);
}

// B-matrix 32x16 fragment where the SOURCE is stored N-major with k contiguous
// (i.e. value = base[n*nstride + k]).  Same two-b128 pattern as A.
__device__ __forceinline__ v16bf frag_bt_ld(const bf16_t* base, int nstride) {
  const int lane = threadIdx.x & 31;
  const int n = lane & 15, half = lane >> 4;
  const bf16_t* p = base + n * nstride + (half << 3);
  v8bf lo = *(const v8bf*)p;
  v8bf hi = *(const v8bf*)(p + 16);
  return __builtin_shufflevector(lo, hi, 0, 1, 2, 3, 4, 5, 6, 7,
                                 8, 9, 10, 11, 12, 13, 14, 15);
}

// B-matrix fragment from a pre-swizzled (fragment-packed) tile:
// tile = 32 lanes x 16 bf16 contiguous -> one 32B load per lane.
__device__ __forceinline__ v16bf frag_b_pk(const bf16_t* packed) {
  const int lane = threadIdx.x & 31;
  return *(const v16bf*)(packed + lane * 16);
}

__device__ __forceinline__ v8f wmma_bf16(v16bf a, v16bf b, v8f c) {
  return __builtin_amdgcn_wmma_f32_16x16x32_bf16(false, a, false, b, (short)0, c,
                                                 false, false);
}

// K index inside a 32-deep tile for packed-B element (e, lane)
__device__ __forceinline__ int pk_k(int e, int lane) {
  return ((e & 8) << 1) + ((lane >> 4) << 3) + (e & 7);
}

// ---------- prep: conv weights -> packed B tiles (K=1024, N=64 per branch) ----------
__global__ void prep_wb_kernel(PtrsF4 cw, bf16_t* __restrict__ wbp) {
  int idx = blockIdx.x * blockDim.x + threadIdx.x;
  if (idx >= 4 * 128 * 512) return;
  int e = idx & 15;
  int lane = (idx >> 4) & 31;
  int tile = (idx >> 9) & 127;  // kt*4 + nt
  int b = idx >> 16;
  int kt = tile >> 2, nt = tile & 3;
  int k = kt * 32 + pk_k(e, lane);  // k = r*64 + i
  int r = k >> 6, i = k & 63;
  int o = nt * 16 + (lane & 15);
  wbp[idx] = (bf16_t)cw.p[b][o * 1024 + i * 16 + r];
}

// ---------- prep: packed wqkv/wo/fcw + forward & inverse permutations ----------
__global__ void prep_misc_kernel(const float* __restrict__ wq, const float* __restrict__ wk,
                                 const float* __restrict__ wv, const float* __restrict__ wo,
                                 const float* __restrict__ fcw, PtrsI4 perms,
                                 bf16_t* __restrict__ wqkvp, bf16_t* __restrict__ wop,
                                 bf16_t* __restrict__ fcwp, int* __restrict__ ip,
                                 int* __restrict__ fperm) {
  int idx = blockIdx.x * blockDim.x + threadIdx.x;
  if (idx < 48 * 512) {  // wqkv: K=64, N=384 -> ktiles=2, ntiles=24
    int e = idx & 15, lane = (idx >> 4) & 31, tile = idx >> 9;
    int kt = tile / 24, nt = tile % 24;
    int k = kt * 32 + pk_k(e, lane);
    int j = nt * 16 + (lane & 15);
    float v = (j < 128) ? wq[k * 128 + j]
            : (j < 256) ? wk[k * 128 + (j - 128)]
                        : wv[k * 128 + (j - 256)];
    wqkvp[idx] = (bf16_t)v;
    return;
  }
  idx -= 48 * 512;
  if (idx < 16 * 512) {  // wo: K=128, N=64 -> ktiles=4, ntiles=4
    int e = idx & 15, lane = (idx >> 4) & 31, tile = idx >> 9;
    int kt = tile >> 2, nt = tile & 3;
    int k = kt * 32 + pk_k(e, lane);
    wop[idx] = (bf16_t)wo[k * 64 + nt * 16 + (lane & 15)];
    return;
  }
  idx -= 16 * 512;
  if (idx < 32 * 512) {  // fcw: K=256, N=64 -> ktiles=8, ntiles=4
    int e = idx & 15, lane = (idx >> 4) & 31, tile = idx >> 9;
    int kt = tile >> 2, nt = tile & 3;
    int k = kt * 32 + pk_k(e, lane);
    fcwp[idx] = (bf16_t)fcw[k * 64 + nt * 16 + (lane & 15)];
    return;
  }
  idx -= 32 * 512;
  if (idx < 2048) { ip[idx] = idx; return; }  // branch 0 inverse: identity
  idx -= 2048;
  if (idx < 3 * 2048) {                        // inverse: ip[perm[j]] = j
    int b = idx >> 11;
    int j = idx & 2047;
    ip[(b + 1) * 2048 + perms.p[b + 1][j]] = j;
    return;
  }
  idx -= 3 * 2048;
  if (idx < 2048) { fperm[idx] = idx; return; }  // branch 0 forward: identity
  idx -= 2048;
  if (idx < 3 * 2048) {                           // forward copy
    int b = idx >> 11;
    int j = idx & 2047;
    fperm[(b + 1) * 2048 + j] = perms.p[b + 1][j];
  }
}

// ---------- conv branch as GEMM: M=128(g) N=64(o) K=1024(r,i) ----------
// Wave-private LDS tiles + in-order DS per wave => no workgroup barriers needed.
__global__ void __launch_bounds__(256) conv_kernel(
    const float* __restrict__ x, const bf16_t* __restrict__ wbp,
    PtrsF4 cb, const int* __restrict__ fperm,
    float* __restrict__ cbrf, bf16_t* __restrict__ cbr16) {
  __shared__ alignas(16) bf16_t lds_a[8][16][64];
  const int bt = blockIdx.x, br = blockIdx.y;
  const int wave = threadIdx.x >> 5, lane = threadIdx.x & 31;
  const int g0 = wave * 16;
  const bf16_t* wbb = wbp + br * 65536;
  // each lane owns one row (lane&15) and one 32-col half (lane>=16)
  const int row = lane & 15, chalf = (lane >> 4) * 32;
  const int* pidx = fperm + br * NN + (g0 + row) * RR;
  bf16_t* dst = &lds_a[wave][row][chalf];
  v8f acc[4];
#pragma unroll
  for (int vt = 0; vt < 4; ++vt) acc[vt] = v8f_zero();

  for (int r = 0; r < RR; ++r) {
    // stage permuted A chunk: 1 index load + one 8-wide clause of b128 loads,
    // then convert+store (keeps loads batched; no per-load full waits)
    const int n = pidx[r];
    const float* src = x + ((size_t)bt * NN + n) * DD + chalf;
    float4 f[8];
#pragma unroll
    for (int t = 0; t < 8; ++t) f[t] = *(const float4*)(src + t * 4);
#pragma unroll
    for (int t = 0; t < 8; ++t) {
      v4bf o;
      o[0] = (bf16_t)f[t].x; o[1] = (bf16_t)f[t].y;
      o[2] = (bf16_t)f[t].z; o[3] = (bf16_t)f[t].w;
      *(v4bf*)(dst + t * 4) = o;
    }
#pragma unroll
    for (int s = 0; s < 2; ++s) {
      v16bf a = frag_a_ld(&lds_a[wave][0][0] + s * 32, 64);
#pragma unroll
      for (int vt = 0; vt < 4; ++vt)
        acc[vt] = wmma_bf16(a, frag_b_pk(wbb + ((r * 2 + s) * 4 + vt) * 512), acc[vt]);
    }
  }
  const float* bias = cb.p[br];
  const size_t base = ((size_t)(br * BT + bt)) * GG * DD;
#pragma unroll
  for (int vt = 0; vt < 4; ++vt) {
#pragma unroll
    for (int v = 0; v < 8; ++v) {
      int m = v + 8 * (lane >> 4);
      int col = vt * 16 + (lane & 15);
      float val = acc[vt][v] + bias[col];
      size_t o = base + (size_t)(g0 + m) * DD + col;
      cbrf[o] = val;
      cbr16[o] = (bf16_t)val;
    }
  }
}

// ---------- QKV projection: M=128 K=64 N=384; V stored transposed ----------
__global__ void __launch_bounds__(256) qkv_kernel(
    const bf16_t* __restrict__ cbr16, const bf16_t* __restrict__ wqkvp,
    bf16_t* __restrict__ qkb, bf16_t* __restrict__ vtb) {
  const int bt = blockIdx.x, br = blockIdx.y;
  const int wave = threadIdx.x >> 5, lane = threadIdx.x & 31;
  const int g0 = wave * 16;
  const size_t abase = ((size_t)(br * BT + bt)) * GG * DD + (size_t)g0 * DD;
  v16bf a0 = frag_a_ld(cbr16 + abase, 64);
  v16bf a1 = frag_a_ld(cbr16 + abase + 32, 64);
  bf16_t* qk = qkb + ((size_t)(br * BT + bt)) * GG * 256;  // [g][q(128)|k(128)]
  bf16_t* vt = vtb + ((size_t)(br * BT + bt)) * HD * GG;   // [d(128)][g(128)]
  for (int nt = 0; nt < 24; ++nt) {
    v8f acc = v8f_zero();
    acc = wmma_bf16(a0, frag_b_pk(wqkvp + (0 * 24 + nt) * 512), acc);
    acc = wmma_bf16(a1, frag_b_pk(wqkvp + (1 * 24 + nt) * 512), acc);
    if (nt < 16) {  // q,k -> row-major [G][256]
#pragma unroll
      for (int v = 0; v < 8; ++v) {
        int m = v + 8 * (lane >> 4);
        qk[(size_t)(g0 + m) * 256 + nt * 16 + (lane & 15)] = (bf16_t)acc[v];
      }
    } else {        // v -> transposed [d][g] so PV B-frags are k-contiguous
#pragma unroll
      for (int v = 0; v < 8; ++v) {
        int m = v + 8 * (lane >> 4);
        int c = (nt - 16) * 16 + (lane & 15);
        vt[(size_t)c * GG + (g0 + m)] = (bf16_t)acc[v];
      }
    }
  }
}

// ---------- attention per (bt, branch, head): S=QK^T, softmax, O=PV ----------
__global__ void __launch_bounds__(256) attn_kernel(
    const bf16_t* __restrict__ qkb, const bf16_t* __restrict__ vtb,
    bf16_t* __restrict__ obuf) {
  __shared__ alignas(16) bf16_t lds_p[8][16][128];
  const int bt = blockIdx.x, br = blockIdx.y, h = blockIdx.z;
  const int wave = threadIdx.x >> 5, lane = threadIdx.x & 31;
  const int g0 = wave * 16;
  const bf16_t* qk = qkb + ((size_t)(br * BT + bt)) * GG * 256;
  const bf16_t* vt = vtb + ((size_t)(br * BT + bt)) * HD * GG;

  v16bf aq0 = frag_a_ld(qk + (size_t)g0 * 256 + h * 64, 256);
  v16bf aq1 = frag_a_ld(qk + (size_t)g0 * 256 + h * 64 + 32, 256);

  v8f accS[8];
#pragma unroll
  for (int nt = 0; nt < 8; ++nt) {
    accS[nt] = v8f_zero();
    // B = K^T : K[g'][d], k(=d) contiguous, n(=g') stride 256
    accS[nt] = wmma_bf16(aq0,
        frag_bt_ld(qk + (size_t)(nt * 16) * 256 + 128 + h * 64, 256), accS[nt]);
    accS[nt] = wmma_bf16(aq1,
        frag_bt_ld(qk + (size_t)(nt * 16) * 256 + 128 + h * 64 + 32, 256), accS[nt]);
  }

  // row softmax: each C row lives on 16 lanes (masks 1/2/4/8 stay in-group)
#pragma unroll
  for (int v = 0; v < 8; ++v) {
    int m = v + 8 * (lane >> 4);
    float mx = -3.0e38f;
#pragma unroll
    for (int nt = 0; nt < 8; ++nt) {
      float sv = accS[nt][v] * 0.125f;  // 1/sqrt(D)
      accS[nt][v] = sv;
      mx = fmaxf(mx, sv);
    }
    mx = fmaxf(mx, __shfl_xor(mx, 1));
    mx = fmaxf(mx, __shfl_xor(mx, 2));
    mx = fmaxf(mx, __shfl_xor(mx, 4));
    mx = fmaxf(mx, __shfl_xor(mx, 8));
    float sum = 0.f;
#pragma unroll
    for (int nt = 0; nt < 8; ++nt) {
      float p = __expf(accS[nt][v] - mx);
      accS[nt][v] = p;
      sum += p;
    }
    sum += __shfl_xor(sum, 1);
    sum += __shfl_xor(sum, 2);
    sum += __shfl_xor(sum, 4);
    sum += __shfl_xor(sum, 8);
    float inv = 1.0f / sum;
#pragma unroll
    for (int nt = 0; nt < 8; ++nt)
      lds_p[wave][m][nt * 16 + (lane & 15)] = (bf16_t)(accS[nt][v] * inv);
  }
  // wave-private LDS region; DS ops are in-order within a wave -> no barrier

  // O = P(16x128) @ V(128x64); V read from vt[d][g] (k contiguous)
  v16bf pa[4];
#pragma unroll
  for (int s = 0; s < 4; ++s) pa[s] = frag_a_ld(&lds_p[wave][0][0] + s * 32, 128);
  bf16_t* ob = obuf + ((size_t)(br * BT + bt)) * GG * HD;
#pragma unroll
  for (int vt4 = 0; vt4 < 4; ++vt4) {
    v8f acc = v8f_zero();
#pragma unroll
    for (int s = 0; s < 4; ++s)
      acc = wmma_bf16(pa[s],
          frag_bt_ld(vt + (size_t)(h * 64 + vt4 * 16) * GG + s * 32, GG), acc);
#pragma unroll
    for (int v = 0; v < 8; ++v) {
      int m = v + 8 * (lane >> 4);
      int col = h * 64 + vt4 * 16 + (lane & 15);
      ob[(size_t)(g0 + m) * HD + col] = (bf16_t)acc[v];
    }
  }
}

// ---------- out-proj + residual + layernorm: M=128 K=128 N=64 ----------
__global__ void __launch_bounds__(256) outproj_kernel(
    const bf16_t* __restrict__ obuf, const bf16_t* __restrict__ wop,
    const float* __restrict__ cbrf, const float* __restrict__ ga,
    const float* __restrict__ ba, bf16_t* __restrict__ abr) {
  const int bt = blockIdx.x, br = blockIdx.y;
  const int wave = threadIdx.x >> 5, lane = threadIdx.x & 31;
  const int g0 = wave * 16;
  const bf16_t* ob = obuf + ((size_t)(br * BT + bt)) * GG * HD + (size_t)g0 * HD;
  v16bf pa[4];
#pragma unroll
  for (int s = 0; s < 4; ++s) pa[s] = frag_a_ld(ob + s * 32, HD);
  v8f acc[4];
#pragma unroll
  for (int vt = 0; vt < 4; ++vt) {
    acc[vt] = v8f_zero();
#pragma unroll
    for (int s = 0; s < 4; ++s)
      acc[vt] = wmma_bf16(pa[s], frag_b_pk(wop + (s * 4 + vt) * 512), acc[vt]);
  }
  const size_t cbase = ((size_t)(br * BT + bt)) * GG * DD;
#pragma unroll
  for (int v = 0; v < 8; ++v) {
    int m = v + 8 * (lane >> 4);
    float vals[4];
    float s1 = 0.f, s2 = 0.f;
#pragma unroll
    for (int vt = 0; vt < 4; ++vt) {
      int col = vt * 16 + (lane & 15);
      float val = acc[vt][v] + cbrf[cbase + (size_t)(g0 + m) * DD + col];
      vals[vt] = val;
      s1 += val;
      s2 += val * val;
    }
    s1 += __shfl_xor(s1, 1); s2 += __shfl_xor(s2, 1);
    s1 += __shfl_xor(s1, 2); s2 += __shfl_xor(s2, 2);
    s1 += __shfl_xor(s1, 4); s2 += __shfl_xor(s2, 4);
    s1 += __shfl_xor(s1, 8); s2 += __shfl_xor(s2, 8);
    float mean = s1 * (1.0f / 64.0f);
    float var = s2 * (1.0f / 64.0f) - mean * mean;
    float rstd = rsqrtf(var + 1e-6f);
#pragma unroll
    for (int vt = 0; vt < 4; ++vt) {
      int col = vt * 16 + (lane & 15);
      float y = (vals[vt] - mean) * rstd * ga[col] + ba[col];
      abr[cbase + (size_t)(g0 + m) * DD + col] = (bf16_t)y;
    }
  }
}

// ---------- final: gather/un-permute + cat@fcw + residual + relu + LN ----------
__global__ void __launch_bounds__(256) final_kernel(
    const bf16_t* __restrict__ abr, const bf16_t* __restrict__ fcwp,
    const int* __restrict__ ip, const float* __restrict__ x,
    const float* __restrict__ g2, const float* __restrict__ b2,
    float* __restrict__ out) {
  __shared__ alignas(16) bf16_t lds_a[8][16][64];
  const int bt = blockIdx.x;
  const int wave = threadIdx.x >> 5, lane = threadIdx.x & 31;
  const int n0 = blockIdx.y * 128 + wave * 16;
  const int row = lane & 15, chalf = (lane >> 4) * 32;
  bf16_t* dst = &lds_a[wave][row][chalf];
  v8f acc[4];
#pragma unroll
  for (int vt = 0; vt < 4; ++vt) acc[vt] = v8f_zero();

  for (int br = 0; br < 4; ++br) {
    const bf16_t* ab = abr + ((size_t)(br * BT + bt)) * GG * DD;
    // 1 index load + 4 b128 copies per lane (lane owns row, 32-col half)
    const int g = ip[br * NN + n0 + row] >> 4;  // argsort(perm)[n] / R
    const bf16_t* src = ab + (size_t)g * DD + chalf;
    v8bf c[4];
#pragma unroll
    for (int t = 0; t < 4; ++t) c[t] = *(const v8bf*)(src + t * 8);
#pragma unroll
    for (int t = 0; t < 4; ++t) *(v8bf*)(dst + t * 8) = c[t];
#pragma unroll
    for (int s = 0; s < 2; ++s) {
      v16bf a = frag_a_ld(&lds_a[wave][0][0] + s * 32, 64);
#pragma unroll
      for (int vt = 0; vt < 4; ++vt)
        acc[vt] = wmma_bf16(a, frag_b_pk(fcwp + ((br * 2 + s) * 4 + vt) * 512), acc[vt]);
    }
  }

#pragma unroll
  for (int v = 0; v < 8; ++v) {
    int m = v + 8 * (lane >> 4);
    size_t rowb = ((size_t)bt * NN + (n0 + m)) * DD;
    float vals[4];
    float s1 = 0.f, s2 = 0.f;
#pragma unroll
    for (int vt = 0; vt < 4; ++vt) {
      int col = vt * 16 + (lane & 15);
      float val = fmaxf(acc[vt][v] + x[rowb + col], 0.0f);  // relu(cat@fcw + x)
      vals[vt] = val;
      s1 += val;
      s2 += val * val;
    }
    s1 += __shfl_xor(s1, 1); s2 += __shfl_xor(s2, 1);
    s1 += __shfl_xor(s1, 2); s2 += __shfl_xor(s2, 2);
    s1 += __shfl_xor(s1, 4); s2 += __shfl_xor(s2, 4);
    s1 += __shfl_xor(s1, 8); s2 += __shfl_xor(s2, 8);
    float mean = s1 * (1.0f / 64.0f);
    float var = s2 * (1.0f / 64.0f) - mean * mean;
    float rstd = rsqrtf(var + 1e-6f);
#pragma unroll
    for (int vt = 0; vt < 4; ++vt) {
      int col = vt * 16 + (lane & 15);
      out[rowb + col] = (vals[vt] - mean) * rstd * g2[col] + b2[col];
    }
  }
}

extern "C" void kernel_launch(void* const* d_in, const int* in_sizes, int n_in,
                              void* d_out, int out_size, void* d_ws, size_t ws_size,
                              hipStream_t stream) {
  (void)in_sizes; (void)n_in; (void)out_size; (void)ws_size;
  const float* x = (const float*)d_in[0];
  // d_in[1] cross_input: unused by reference
  PtrsI4 perms = {{nullptr, (const int*)d_in[2], (const int*)d_in[3], (const int*)d_in[4]}};
  PtrsF4 cw = {{(const float*)d_in[5], (const float*)d_in[6],
                (const float*)d_in[7], (const float*)d_in[8]}};
  PtrsF4 cb = {{(const float*)d_in[9], (const float*)d_in[10],
                (const float*)d_in[11], (const float*)d_in[12]}};
  const float* wq  = (const float*)d_in[13];
  const float* wk  = (const float*)d_in[14];
  const float* wv  = (const float*)d_in[15];
  const float* wo  = (const float*)d_in[16];
  const float* ga  = (const float*)d_in[17];
  const float* ba  = (const float*)d_in[18];
  const float* fcw = (const float*)d_in[19];
  const float* g2  = (const float*)d_in[20];
  const float* b2  = (const float*)d_in[21];

  char* wsp = (char*)d_ws;
  size_t off = 0;
  auto carve = [&](size_t bytes) -> void* {
    void* p = wsp + off;
    off += (bytes + 255) & ~(size_t)255;
    return p;
  };
  bf16_t* wbp   = (bf16_t*)carve((size_t)4 * 128 * 512 * 2);   // packed conv weights
  bf16_t* wqkvp = (bf16_t*)carve((size_t)48 * 512 * 2);        // packed [wq|wk|wv]
  bf16_t* wop   = (bf16_t*)carve((size_t)16 * 512 * 2);        // packed wo
  bf16_t* fcwp  = (bf16_t*)carve((size_t)32 * 512 * 2);        // packed fcw
  int*    ip    = (int*)   carve((size_t)4 * NN * 4);          // inverse perms
  int*    fperm = (int*)   carve((size_t)4 * NN * 4);          // forward perms (id for br0)
  float*  cbrf  = (float*) carve((size_t)4 * BT * GG * DD * 4);
  bf16_t* cbr16 = (bf16_t*)carve((size_t)4 * BT * GG * DD * 2);
  bf16_t* qkb   = (bf16_t*)carve((size_t)4 * BT * GG * 256 * 2);  // [g][q|k]
  bf16_t* vtb   = (bf16_t*)carve((size_t)4 * BT * HD * GG * 2);   // v transposed
  bf16_t* obuf  = (bf16_t*)carve((size_t)4 * BT * GG * HD * 2);
  bf16_t* abr   = (bf16_t*)carve((size_t)4 * BT * GG * DD * 2);

  prep_wb_kernel<<<1024, 256, 0, stream>>>(cw, wbp);
  prep_misc_kernel<<<256, 256, 0, stream>>>(wq, wk, wv, wo, fcw, perms,
                                            wqkvp, wop, fcwp, ip, fperm);
  conv_kernel<<<dim3(BT, 4), 256, 0, stream>>>(x, wbp, cb, fperm, cbrf, cbr16);
  qkv_kernel<<<dim3(BT, 4), 256, 0, stream>>>(cbr16, wqkvp, qkb, vtb);
  attn_kernel<<<dim3(BT, 4, 2), 256, 0, stream>>>(qkb, vtb, obuf);
  outproj_kernel<<<dim3(BT, 4), 256, 0, stream>>>(obuf, wop, cbrf, ga, ba, abr);
  final_kernel<<<dim3(BT, 16), 256, 0, stream>>>(abr, fcwp, ip, x, g2, b2,
                                                 (float*)d_out);
}
